// Transformer_16784732193201
// MI455X (gfx1250) — matrix operand order repaired and checked by reading, not compiled, not run on hardware
//
#include <hip/hip_runtime.h>
#include <hip/hip_bf16.h>

// MI455X / gfx1250, wave32, WMMA 16x16x32 f16.

typedef _Float16 v8h  __attribute__((ext_vector_type(8)));
typedef _Float16 v16h __attribute__((ext_vector_type(16)));
typedef float    v8f  __attribute__((ext_vector_type(8)));

#define SHUF16(lo, hi) __builtin_shufflevector((lo), (hi), 0,1,2,3,4,5,6,7,8,9,10,11,12,13,14,15)

// ---------------- elementwise helpers ----------------

__global__ __launch_bounds__(256) void k_f32_to_f16(const float* __restrict__ s,
                                                    _Float16* __restrict__ d, int n) {
  int i = blockIdx.x * 256 + threadIdx.x;
  if (i < n) d[i] = (_Float16)s[i];
}

__global__ __launch_bounds__(256) void k_addpos(const float* __restrict__ x,
                                                const float* __restrict__ pe,
                                                float* __restrict__ X, int total, int per) {
  int i = blockIdx.x * 256 + threadIdx.x;
  if (i < total) X[i] = x[i] + pe[i % per];
}

__global__ __launch_bounds__(256) void k_bn(const float* __restrict__ X,
                                            const float* __restrict__ g,
                                            const float* __restrict__ b,
                                            _Float16* __restrict__ out, int total) {
  int i = blockIdx.x * 256 + threadIdx.x;
  if (i < total) {
    int c = i & 255;
    // x / sqrt(1 + 1e-5) * g + b
    out[i] = (_Float16)(X[i] * 0.99999500003749969f * g[c] + b[c]);
  }
}

// LayerNorm over D=256: one ROW PER WAVE (8 lanes' worth = 8 f32 each), butterfly
// shuffle reductions, no LDS, no block barriers. Unbiased var, a*(x-mean)/(sqrt(var)+eps)+b.
__global__ __launch_bounds__(256) void k_ln(const float* __restrict__ X,
                                            const float* __restrict__ ga,
                                            const float* __restrict__ be,
                                            _Float16* __restrict__ out) {
  const int wave = threadIdx.x >> 5, lane = threadIdx.x & 31;
  size_t row = (size_t)blockIdx.x * 8 + wave;
  const float* p = X + row * 256 + lane * 8;
  float4 va = *(const float4*)p;
  float4 vb = *(const float4*)(p + 4);
  float x[8] = {va.x, va.y, va.z, va.w, vb.x, vb.y, vb.z, vb.w};
  float s = 0.f;
#pragma unroll
  for (int j = 0; j < 8; ++j) s += x[j];
#pragma unroll
  for (int m = 16; m >= 1; m >>= 1) s += __shfl_xor(s, m, 32);
  float mean = s * (1.0f / 256.0f);
  float q = 0.f;
#pragma unroll
  for (int j = 0; j < 8; ++j) { x[j] -= mean; q += x[j] * x[j]; }
#pragma unroll
  for (int m = 16; m >= 1; m >>= 1) q += __shfl_xor(q, m, 32);
  float inv = 1.0f / (sqrtf(q * (1.0f / 255.0f)) + 1e-6f);
  const float* gp = ga + lane * 8;
  const float* bp = be + lane * 8;
  float4 g0 = *(const float4*)gp, g1 = *(const float4*)(gp + 4);
  float4 b0 = *(const float4*)bp, b1 = *(const float4*)(bp + 4);
  float gg[8] = {g0.x, g0.y, g0.z, g0.w, g1.x, g1.y, g1.z, g1.w};
  float bb[8] = {b0.x, b0.y, b0.z, b0.w, b1.x, b1.y, b1.z, b1.w};
  v8h o;
#pragma unroll
  for (int j = 0; j < 8; ++j) o[j] = (_Float16)(gg[j] * x[j] * inv + bb[j]);
  *(v8h*)(out + row * 256 + lane * 8) = o;
}

// ---------------- WMMA GEMM: C[M,N] = A[M,K] * W[N,K]^T + bias ----------------
// 256 threads = 8 waves; wave = (mw in 0..1, nw in 0..3).
// Wave tile = 32 rows x 64 cols: 2 M-subtiles x 4 N-tiles = 8 v8f accumulators;
// each B fragment feeds 2 WMMAs, each A fragment feeds 4 -> 12 b128 loads per 8 WMMAs.
// Block tile = 64 rows x 256 cols; grid = (Mrows/64, ceil(N/256)).
// MODE 0: f16 out (optional ReLU). MODE 1: f32 out += residual. MODE 2: f32 out.
template <int MODE, bool RELU, bool NBOUND>
__global__ __launch_bounds__(256) void k_gemm(const _Float16* __restrict__ A,
                                              const _Float16* __restrict__ W,
                                              const float* __restrict__ bias,
                                              float* __restrict__ outF,
                                              _Float16* __restrict__ outH,
                                              const float* __restrict__ res,
                                              int N, int K) {
  const int lane = threadIdx.x & 31;
  const int wave = threadIdx.x >> 5;
  const int mw = wave & 1, nw = wave >> 1;
  const int rowBase = blockIdx.x * 64 + mw * 32;
  const int colBase = blockIdx.y * 256 + nw * 64;
  const int l16 = lane & 15;
  const int ko = (lane < 16) ? 0 : 8;   // A-frag K offset (ISA 16-bit A 16x32 layout)
  const int kb = (lane < 16) ? 0 : 16;  // unused; B-frag uses ko, same k order as A

  v8f acc[2][4] = {};
  const _Float16* aRow0 = A + (size_t)(rowBase + l16) * K;
  const _Float16* aRow1 = aRow0 + (size_t)16 * K;
  for (int k0 = 0; k0 < K; k0 += 32) {
    v8h a0lo = *(const v8h*)(aRow0 + k0 + ko);
    v8h a0hi = *(const v8h*)(aRow0 + k0 + 16 + ko);
    v8h a1lo = *(const v8h*)(aRow1 + k0 + ko);
    v8h a1hi = *(const v8h*)(aRow1 + k0 + 16 + ko);
    v16h af0 = SHUF16(a0lo, a0hi);
    v16h af1 = SHUF16(a1lo, a1hi);
#pragma unroll
    for (int t = 0; t < 4; ++t) {
      int n = colBase + t * 16 + l16;
      v16h bf = {};
      if (!NBOUND || n < N) {
        const _Float16* wRow = W + (size_t)n * K + k0 + ko;
        v8h blo = *(const v8h*)(wRow);
        v8h bhi = *(const v8h*)(wRow + 16);
        bf = SHUF16(blo, bhi);
      }
      acc[0][t] = __builtin_amdgcn_wmma_f32_16x16x32_f16(false, af0, false, bf,
                                                         (short)0, acc[0][t], false, false);
      acc[1][t] = __builtin_amdgcn_wmma_f32_16x16x32_f16(false, af1, false, bf,
                                                         (short)0, acc[1][t], false, false);
    }
  }
  const int rowOff = (lane >> 4) * 8;  // C layout: vgpr v -> M = v + 8*(lane>=16)
#pragma unroll
  for (int s = 0; s < 2; ++s) {
#pragma unroll
    for (int t = 0; t < 4; ++t) {
      int n = colBase + t * 16 + l16;
      if (NBOUND && n >= N) continue;
      float bv = bias[n];
#pragma unroll
      for (int v = 0; v < 8; ++v) {
        size_t idx = (size_t)(rowBase + s * 16 + rowOff + v) * N + n;
        float val = acc[s][t][v] + bv;
        if (RELU) val = fmaxf(val, 0.0f);
        if (MODE == 0)      outH[idx] = (_Float16)val;
        else if (MODE == 1) outF[idx] = val + res[idx];
        else                outF[idx] = val;
      }
    }
  }
}

// ---------------- strided sparse attention ----------------
// One wave per (b,h,m) stream: TM=27 positions (pad 32), DK=32 (== WMMA K!).
// scores = Q K^T / sqrt(32) (4 wmma), softmax over 27 keys in LDS, O = P V (4 wmma).
__global__ __launch_bounds__(128) void k_attn(const _Float16* __restrict__ Q,
                                              const _Float16* __restrict__ Km,
                                              const _Float16* __restrict__ V,
                                              _Float16* __restrict__ O) {
  __shared__ __align__(16) float    sS[4][32][32];
  __shared__ __align__(16) _Float16 sP[4][32][32];
  const int lane = threadIdx.x & 31;
  const int w    = threadIdx.x >> 5;
  int idx = blockIdx.x * 4 + w;               // (b,h,m) flat, B*H*M = 24576
  int b = idx / 24, rem = idx % 24, h = rem / 3, m = rem % 3;
  size_t base = ((size_t)b * 81 + m) * 256 + (size_t)h * 32;  // pos g -> +g*768
  const int l16 = lane & 15;
  const int ko = (lane < 16) ? 0 : 8;
  const int kb = (lane < 16) ? 0 : 16;
  const int rowOff = (lane >> 4) * 8;

  v16h aQ[2], bK[2];
#pragma unroll
  for (int mi = 0; mi < 2; ++mi) {
    int q = mi * 16 + l16;
    v16h f = {};
    if (q < 27) {
      const _Float16* p = Q + base + (size_t)q * 768;
      v8h lo = *(const v8h*)(p + ko);
      v8h hi = *(const v8h*)(p + 16 + ko);
      f = SHUF16(lo, hi);
    }
    aQ[mi] = f;
  }
#pragma unroll
  for (int ki = 0; ki < 2; ++ki) {
    int kp = ki * 16 + l16;
    v16h f = {};
    if (kp < 27) {
      const _Float16* p = Km + base + (size_t)kp * 768 + ko;
      v8h lo = *(const v8h*)(p);
      v8h hi = *(const v8h*)(p + 16);
      f = SHUF16(lo, hi);
    }
    bK[ki] = f;
  }
#pragma unroll
  for (int mi = 0; mi < 2; ++mi)
#pragma unroll
    for (int ki = 0; ki < 2; ++ki) {
      v8f s = {};
      s = __builtin_amdgcn_wmma_f32_16x16x32_f16(false, aQ[mi], false, bK[ki],
                                                 (short)0, s, false, false);
#pragma unroll
      for (int v = 0; v < 8; ++v)
        sS[w][mi * 16 + rowOff + v][ki * 16 + l16] = s[v] * 0.17677669529663687f;
    }
  __syncthreads();
  if (lane < 27) {
    float mx = -1e30f;
    for (int k = 0; k < 27; ++k) mx = fmaxf(mx, sS[w][lane][k]);
    float sum = 0.f;
    for (int k = 0; k < 27; ++k) {
      float e = __expf(sS[w][lane][k] - mx);
      sum += e;
      sS[w][lane][k] = e;
    }
    float inv = 1.0f / sum;
    for (int k = 0; k < 27; ++k) sP[w][lane][k] = (_Float16)(sS[w][lane][k] * inv);
    for (int k = 27; k < 32; ++k) sP[w][lane][k] = (_Float16)0.f;
  } else {
    for (int k = 0; k < 32; ++k) sP[w][lane][k] = (_Float16)0.f;
  }
  __syncthreads();
  v16h aP[2], bV[2];
#pragma unroll
  for (int mi = 0; mi < 2; ++mi) {
    const _Float16* p = &sP[w][mi * 16 + l16][0];
    v8h lo = *(const v8h*)(p + ko);
    v8h hi = *(const v8h*)(p + 16 + ko);
    aP[mi] = SHUF16(lo, hi);
  }
#pragma unroll
  for (int nt = 0; nt < 2; ++nt) {
    int d = nt * 16 + l16;
    v16h f = {};
#pragma unroll
    for (int e = 0; e < 16; ++e) {
      int k = (e < 8) ? (ko + e) : (16 + ko + (e - 8));
      if (k < 27) f[e] = V[base + (size_t)k * 768 + d];
    }
    bV[nt] = f;
  }
#pragma unroll
  for (int mi = 0; mi < 2; ++mi)
#pragma unroll
    for (int nt = 0; nt < 2; ++nt) {
      v8f o = {};
      o = __builtin_amdgcn_wmma_f32_16x16x32_f16(false, aP[mi], false, bV[nt],
                                                 (short)0, o, false, false);
#pragma unroll
      for (int v = 0; v < 8; ++v) {
        int q = mi * 16 + rowOff + v;
        if (q < 27)
          O[base + (size_t)q * 768 + nt * 16 + l16] = (_Float16)o[v];
      }
    }
}

// ---------------- host orchestration ----------------

extern "C" void kernel_launch(void* const* d_in, const int* in_sizes, int n_in,
                              void* d_out, int out_size, void* d_ws, size_t ws_size,
                              hipStream_t stream) {
  (void)in_sizes; (void)n_in; (void)out_size; (void)ws_size;
  const float* x      = (const float*)d_in[0];
  const float* pos    = (const float*)d_in[1];
  const float* ln1_a  = (const float*)d_in[2];
  const float* ln1_b  = (const float*)d_in[3];
  const float* Wq     = (const float*)d_in[4];
  const float* bq     = (const float*)d_in[5];
  const float* Wk     = (const float*)d_in[6];
  const float* bk     = (const float*)d_in[7];
  const float* Wv     = (const float*)d_in[8];
  const float* bv     = (const float*)d_in[9];
  const float* Wo     = (const float*)d_in[10];
  const float* bo     = (const float*)d_in[11];
  const float* ln2_a  = (const float*)d_in[12];
  const float* ln2_b  = (const float*)d_in[13];
  const float* W1     = (const float*)d_in[14];
  const float* b1     = (const float*)d_in[15];
  const float* W2     = (const float*)d_in[16];
  const float* b2     = (const float*)d_in[17];
  const float* bn_g   = (const float*)d_in[18];
  const float* bn_b   = (const float*)d_in[19];
  const float* conv_w = (const float*)d_in[20];
  const float* conv_b = (const float*)d_in[21];

  const size_t R = 82944;  // B*T
  char* base = (char*)d_ws;
  float*    X   = (float*)base;                               // f32 [R,256]
  _Float16* A16 = (_Float16*)(base + R * 256 * 4);            // f16 [R,256] staging
  _Float16* Q16 = (_Float16*)(base + R * 256 * 4 + R * 256 * 2);
  _Float16* K16 = Q16 + R * 256;
  _Float16* V16 = K16 + R * 256;
  _Float16* H16 = Q16;                                        // [R,512] spans Q16+K16
  _Float16* W16 = V16 + R * 256;                              // f16 weights

  auto cvt = [&](const float* s, _Float16* dst, int n) {
    k_f32_to_f16<<<(n + 255) / 256, 256, 0, stream>>>(s, dst, n);
  };
  const size_t LS = 524288;  // f16 elements per layer of weights
  for (int i = 0; i < 3; ++i) {
    cvt(Wq + (size_t)i * 65536,  W16 + i * LS,          65536);
    cvt(Wk + (size_t)i * 65536,  W16 + i * LS + 65536,  65536);
    cvt(Wv + (size_t)i * 65536,  W16 + i * LS + 131072, 65536);
    cvt(Wo + (size_t)i * 65536,  W16 + i * LS + 196608, 65536);
    cvt(W1 + (size_t)i * 131072, W16 + i * LS + 262144, 131072);
    cvt(W2 + (size_t)i * 131072, W16 + i * LS + 393216, 131072);
  }
  _Float16* Wc16 = W16 + 3 * LS;
  cvt(conv_w, Wc16, 51 * 256);

  int total = (int)(R * 256);
  k_addpos<<<(total + 255) / 256, 256, 0, stream>>>(x, pos, X, total, 81 * 256);

  for (int i = 0; i < 3; ++i) {
    _Float16* Wl = W16 + i * LS;
    k_ln<<<(uint32_t)(R / 8), 256, 0, stream>>>(X, ln1_a + i * 256, ln1_b + i * 256, A16);
    k_gemm<0, false, false><<<dim3(R / 64, 1), 256, 0, stream>>>(A16, Wl,          bq + i * 256, nullptr, Q16, nullptr, 256, 256);
    k_gemm<0, false, false><<<dim3(R / 64, 1), 256, 0, stream>>>(A16, Wl + 65536,  bk + i * 256, nullptr, K16, nullptr, 256, 256);
    k_gemm<0, false, false><<<dim3(R / 64, 1), 256, 0, stream>>>(A16, Wl + 131072, bv + i * 256, nullptr, V16, nullptr, 256, 256);
    k_attn<<<24576 / 4, 128, 0, stream>>>(Q16, K16, V16, A16);
    k_gemm<1, false, false><<<dim3(R / 64, 1), 256, 0, stream>>>(A16, Wl + 196608, bo + i * 256, X, nullptr, X, 256, 256);
    k_ln<<<(uint32_t)(R / 8), 256, 0, stream>>>(X, ln2_a + i * 256, ln2_b + i * 256, A16);
    k_gemm<0, true,  false><<<dim3(R / 64, 2), 256, 0, stream>>>(A16, Wl + 262144, b1 + i * 512, nullptr, H16, nullptr, 512, 256);
    k_gemm<1, false, false><<<dim3(R / 64, 1), 256, 0, stream>>>(H16, Wl + 393216, b2 + i * 256, X, nullptr, X, 256, 512);
  }

  k_bn<<<(total + 255) / 256, 256, 0, stream>>>(X, bn_g, bn_b, A16, total);
  k_gemm<2, false, true><<<dim3(R / 64, 1), 256, 0, stream>>>(A16, Wc16, conv_b,
                                                              (float*)d_out, nullptr, nullptr,
                                                              51, 256);
}
